// SelfAttention_86440511799870
// MI455X (gfx1250) — matrix operand order repaired and checked by reading, not compiled, not run on hardware
//
#include <hip/hip_runtime.h>
#include <hip/hip_bf16.h>

typedef __attribute__((ext_vector_type(16))) _Float16 v16h;
typedef __attribute__((ext_vector_type(8)))  _Float16 v8h;
typedef __attribute__((ext_vector_type(4)))  _Float16 v4h;
typedef __attribute__((ext_vector_type(8)))  float    v8f;

#define BQ   4
#define TT   2048
#define DD   1024
#define HH   16
#define HDIM 64
#define MTOT (BQ*TT)   // 8192

// ---------- helpers ----------
static __device__ inline v16h mk16(v8h lo, v8h hi) {
  v16h r;
#pragma unroll
  for (int i = 0; i < 8; i++) { r[i] = lo[i]; r[8 + i] = hi[i]; }
  return r;
}

// A-fragment (16x32 f16): lane row = lane&15; K = lg*8..+7 and 16+lg*8..+7
static __device__ inline v16h ldafrag(const _Float16* rowp, int lg) {
  v8h lo = *(const v8h*)(rowp + lg * 8);
  v8h hi = *(const v8h*)(rowp + lg * 8 + 16);
  return mk16(lo, hi);
}

// B-fragment (32x16 f16): lane col = lane&15; K = lg*8..+7 and 16+lg*8..+7
static __device__ inline v16h ldbfrag(const _Float16* colp, int lg) {
  v8h lo = *(const v8h*)(colp + lg * 8);
  v8h hi = *(const v8h*)(colp + lg * 8 + 16);
  return mk16(lo, hi);
}

static __device__ inline v8f wmma_f16(v16h a, v16h b, v8f c) {
  return __builtin_amdgcn_wmma_f32_16x16x32_f16(false, a, false, b, (short)0, c,
                                                false, false);
}

// ---------- conversion kernels ----------
__global__ void cvt_f16x4(const float* __restrict__ in, _Float16* __restrict__ out, int n4) {
  int i = blockIdx.x * blockDim.x + threadIdx.x;
  if (i < n4) {
    float4 v = ((const float4*)in)[i];
    v4h h; h[0] = (_Float16)v.x; h[1] = (_Float16)v.y; h[2] = (_Float16)v.z; h[3] = (_Float16)v.w;
    ((v4h*)out)[i] = h;
  }
}

// Tiled transpose + downconvert: Wt[n*K + k] = W[k*N + n]; coalesced both ways.
__global__ void __launch_bounds__(256)
cvt_wT(const float* __restrict__ W, _Float16* __restrict__ Wt, int K, int N) {
  __shared__ _Float16 tile[32][33];
  const int tx = threadIdx.x & 31;
  const int ty = threadIdx.x >> 5;           // 0..7
  const int k0 = blockIdx.x * 32;
  const int n0 = blockIdx.y * 32;
#pragma unroll
  for (int i = 0; i < 4; i++) {
    int k = ty + i * 8;
    tile[k][tx] = (_Float16)W[(size_t)(k0 + k) * N + n0 + tx];
  }
  __syncthreads();
#pragma unroll
  for (int i = 0; i < 4; i++) {
    int n = ty + i * 8;
    Wt[(size_t)(n0 + n) * K + k0 + tx] = tile[tx][n];
  }
}

// ---------- tiled WMMA GEMM: out = A[M,K] * Bt[N,K]^T + bias ----------
// MODE 0: f16 [M,N] out   MODE 1: f16 V-transposed [B,H,HD,T]   MODE 2: f32 [M,N]
template <int MODE>
__global__ void __launch_bounds__(256)
gemm_f16(const _Float16* __restrict__ A, const _Float16* __restrict__ Bt,
         const float* __restrict__ bias, void* __restrict__ outp,
         int M, int N, int K) {
  __shared__ _Float16 As[128][40];
  __shared__ _Float16 Bs[128][40];
  const int tid = threadIdx.x;
  const int wave = tid >> 5;
  const int lane = tid & 31;
  const int lh = lane & 15;
  const int lg = lane >> 4;
  const int m0 = blockIdx.x * 128;
  const int n0 = blockIdx.y * 128;

  v8f zero = {};
  v8f acc[8];
#pragma unroll
  for (int i = 0; i < 8; i++) acc[i] = zero;

  for (int k0 = 0; k0 < K; k0 += 32) {
    // stage 128x32 A tile and 128x32 Bt tile; loads issued before the barrier
    v8h a_ld[2], b_ld[2];
#pragma unroll
    for (int i = 0; i < 2; i++) {
      int c = tid + 256 * i;
      int r = c >> 2;
      int co = (c & 3) * 8;
      a_ld[i] = *(const v8h*)&A[(size_t)(m0 + r) * K + k0 + co];
      b_ld[i] = *(const v8h*)&Bt[(size_t)(n0 + r) * K + k0 + co];
      if (k0 + 32 < K) {
        __builtin_prefetch(&A[(size_t)(m0 + r) * K + k0 + 32 + co], 0, 0);
        __builtin_prefetch(&Bt[(size_t)(n0 + r) * K + k0 + 32 + co], 0, 0);
      }
    }
    __syncthreads();
#pragma unroll
    for (int i = 0; i < 2; i++) {
      int c = tid + 256 * i;
      int r = c >> 2;
      int co = (c & 3) * 8;
      *(v8h*)&As[r][co] = a_ld[i];
      *(v8h*)&Bs[r][co] = b_ld[i];
    }
    __syncthreads();

    v16h af = ldafrag(&As[wave * 16 + lh][0], lg);
#pragma unroll
    for (int nt = 0; nt < 8; nt++) {
      v16h bf = ldbfrag(&Bs[nt * 16 + lh][0], lg);
      acc[nt] = wmma_f16(af, bf, acc[nt]);
    }
  }

#pragma unroll
  for (int nt = 0; nt < 8; nt++) {
#pragma unroll
    for (int j = 0; j < 8; j++) {
      int row = m0 + wave * 16 + j + (lg << 3);
      int col = n0 + nt * 16 + lh;
      float v = acc[nt][j] + bias[col];
      if (MODE == 2) {
        ((float*)outp)[(size_t)row * N + col] = v;
      } else if (MODE == 0) {
        ((_Float16*)outp)[(size_t)row * N + col] = (_Float16)v;
      } else {
        int b = row >> 11, t = row & (TT - 1);
        int h = col >> 6, hd = col & (HDIM - 1);
        ((_Float16*)outp)[(((size_t)b * HH + h) * HDIM + hd) * TT + t] = (_Float16)v;
      }
    }
  }
}

// ---------- flash attention: 8 waves / block, shared K/V staging in LDS ----------
__global__ void __launch_bounds__(256)
attn_kernel(const _Float16* __restrict__ Q,   // [B*T, D], pre-scale applied to frags
            const _Float16* __restrict__ Km,  // [B*T, D]
            const _Float16* __restrict__ Vt,  // [B,H,HD,T]
            const int* __restrict__ mask,     // [T]
            _Float16* __restrict__ ctx) {     // [B*T, D]
  __shared__ _Float16 Ks[32][72];      // keys x hd   (pad -> conflict-free frags)
  __shared__ _Float16 Vs[64][40];      // hd x keys
  __shared__ _Float16 Pl[8][16][40];   // per-wave P relayout buffer

  const int tid = threadIdx.x;
  const int wave = tid >> 5;
  const int lane = tid & 31;
  const int lh = lane & 15;
  const int lg = lane >> 4;

  const int bid = blockIdx.x;          // B*H*(T/128)
  const int qt = bid & 15;
  const int h = (bid >> 4) & 15;
  const int b = bid >> 8;
  const int q0 = qt * 128 + wave * 16;

  // Q fragments, pre-scaled by 1/sqrt(HD) = 0.125 (exact in f16)
  const _Float16* qbase = Q + ((size_t)b * TT + q0 + lh) * DD + h * HDIM;
  v16h qa0 = ldafrag(qbase, lg);
  v16h qa1 = ldafrag(qbase + 32, lg);
#pragma unroll
  for (int i = 0; i < 16; i++) { qa0[i] *= (_Float16)0.125f; qa1[i] *= (_Float16)0.125f; }

  v8f zero = {};
  v8f o0 = zero, o1 = zero, o2 = zero, o3 = zero;
  float mrow[8], lrow[8];
  const float NEG_INF = -__builtin_inff();
#pragma unroll
  for (int j = 0; j < 8; j++) { mrow[j] = NEG_INF; lrow[j] = 0.f; }

  const _Float16* vhead = Vt + ((size_t)b * HH + h) * HDIM * TT;
  const _Float16* kbase = Km + ((size_t)b * TT) * DD + h * HDIM;

  // staging coords (256 threads move 32x64 K-chunk and 64x32 V-chunk)
  const int kr = tid >> 3, kco = (tid & 7) * 8;   // Ks[kr][kco..+7]
  const int vr = tid >> 2, vco = (tid & 3) * 8;   // Vs[vr][vco..+7]

  for (int kt = 0; kt < TT; kt += 32) {
    // issue global loads early, store to LDS after the barrier
    v8h kv = *(const v8h*)&kbase[(size_t)(kt + kr) * DD + kco];
    v8h vv = *(const v8h*)&vhead[(size_t)vr * TT + kt + vco];
    if (kt + 32 < TT) {
      __builtin_prefetch(&kbase[(size_t)(kt + 32 + kr) * DD + kco], 0, 0);
      __builtin_prefetch(&vhead[(size_t)vr * TT + kt + 32 + vco], 0, 0);
    }
    __syncthreads();   // all waves done reading previous Ks/Vs
    *(v8h*)&Ks[kr][kco] = kv;
    *(v8h*)&Vs[vr][vco] = vv;
    __syncthreads();

    // ---- S = (Q*scale) * K^T  (two 16x16 key tiles) ----
    const _Float16* kc0 = &Ks[lh][0];
    const _Float16* kc1 = &Ks[16 + lh][0];
    v8f sc0 = zero, sc1 = zero;
    sc0 = wmma_f16(qa0, ldbfrag(kc0, lg), sc0);
    sc0 = wmma_f16(qa1, ldbfrag(kc0 + 32, lg), sc0);
    sc1 = wmma_f16(qa0, ldbfrag(kc1, lg), sc1);
    sc1 = wmma_f16(qa1, ldbfrag(kc1 + 32, lg), sc1);

    const int msk0 = mask[kt + lh];
    const int msk1 = mask[kt + 16 + lh];

    float p0[8], p1[8], alpha[8];
#pragma unroll
    for (int j = 0; j < 8; j++) {
      float s0 = msk0 ? sc0[j] : NEG_INF;
      float s1 = msk1 ? sc1[j] : NEG_INF;
      float mx = fmaxf(s0, s1);
#pragma unroll
      for (int off = 1; off < 16; off <<= 1) mx = fmaxf(mx, __shfl_xor(mx, off));
      float mnew = fmaxf(mrow[j], mx);
      alpha[j] = __expf(mrow[j] - mnew);
      mrow[j] = mnew;
      p0[j] = __expf(s0 - mnew);
      p1[j] = __expf(s1 - mnew);
      float rs = p0[j] + p1[j];
#pragma unroll
      for (int off = 1; off < 16; off <<= 1) rs += __shfl_xor(rs, off);
      lrow[j] = lrow[j] * alpha[j] + rs;
    }

    // P: C-layout -> A-fragment layout via per-wave LDS bounce
#pragma unroll
    for (int j = 0; j < 8; j++) {
      int r = j + (lg << 3);
      Pl[wave][r][lh] = (_Float16)p0[j];
      Pl[wave][r][16 + lh] = (_Float16)p1[j];
    }
    asm volatile("" ::: "memory");   // same-wave DS is in-order; block compiler reorder
    v16h pa = ldafrag(&Pl[wave][lh][0], lg);

    // rescale running output, then O += P * V
#pragma unroll
    for (int j = 0; j < 8; j++) {
      o0[j] *= alpha[j]; o1[j] *= alpha[j]; o2[j] *= alpha[j]; o3[j] *= alpha[j];
    }
    o0 = wmma_f16(pa, ldbfrag(&Vs[0 * 16 + lh][0], lg), o0);
    o1 = wmma_f16(pa, ldbfrag(&Vs[1 * 16 + lh][0], lg), o1);
    o2 = wmma_f16(pa, ldbfrag(&Vs[2 * 16 + lh][0], lg), o2);
    o3 = wmma_f16(pa, ldbfrag(&Vs[3 * 16 + lh][0], lg), o3);
  }

#pragma unroll
  for (int j = 0; j < 8; j++) {
    float inv = 1.0f / lrow[j];
    int row = q0 + j + (lg << 3);
    _Float16* cp = ctx + ((size_t)b * TT + row) * DD + h * HDIM + lh;
    cp[0]  = (_Float16)(o0[j] * inv);
    cp[16] = (_Float16)(o1[j] * inv);
    cp[32] = (_Float16)(o2[j] * inv);
    cp[48] = (_Float16)(o3[j] * inv);
  }
}

// ---------- host side ----------
extern "C" void kernel_launch(void* const* d_in, const int* in_sizes, int n_in,
                              void* d_out, int out_size, void* d_ws, size_t ws_size,
                              hipStream_t stream) {
  const float* embed = (const float*)d_in[0];
  const int* mask = (const int*)d_in[1];
  const float* Wq = (const float*)d_in[2];
  const float* bq = (const float*)d_in[3];
  const float* Wk = (const float*)d_in[4];
  const float* bk = (const float*)d_in[5];
  const float* Wv = (const float*)d_in[6];
  const float* bv = (const float*)d_in[7];
  const float* Wo = (const float*)d_in[8];
  const float* bo = (const float*)d_in[9];
  float* out = (float*)d_out;

  char* ws = (char*)d_ws;
  size_t off = 0;
  auto carve = [&](size_t bytes) -> char* {
    char* p = ws + off;
    off += (bytes + 255) & ~(size_t)255;
    return p;
  };
  _Float16* X16 = (_Float16*)carve((size_t)MTOT * DD * 2);
  _Float16* WqT = (_Float16*)carve((size_t)DD * DD * 2);
  _Float16* WkT = (_Float16*)carve((size_t)DD * DD * 2);
  _Float16* WvT = (_Float16*)carve((size_t)DD * DD * 2);
  _Float16* WoT = (_Float16*)carve((size_t)DD * DD * 2);
  _Float16* Q16 = (_Float16*)carve((size_t)MTOT * DD * 2);
  _Float16* K16 = (_Float16*)carve((size_t)MTOT * DD * 2);
  _Float16* Vt16 = (_Float16*)carve((size_t)MTOT * DD * 2);
  _Float16* ctx16 = X16;  // X16 dead after V projection; reuse for ctx

  {
    int n4 = MTOT * DD / 4;
    cvt_f16x4<<<(n4 + 255) / 256, 256, 0, stream>>>(embed, X16, n4);
  }
  {
    dim3 tgrid(DD / 32, DD / 32);
    cvt_wT<<<tgrid, 256, 0, stream>>>(Wq, WqT, DD, DD);
    cvt_wT<<<tgrid, 256, 0, stream>>>(Wk, WkT, DD, DD);
    cvt_wT<<<tgrid, 256, 0, stream>>>(Wv, WvT, DD, DD);
    cvt_wT<<<tgrid, 256, 0, stream>>>(Wo, WoT, DD, DD);
  }

  dim3 ggrid(MTOT / 128, DD / 128);
  gemm_f16<0><<<ggrid, 256, 0, stream>>>(X16, WqT, bq, Q16, MTOT, DD, DD);
  gemm_f16<0><<<ggrid, 256, 0, stream>>>(X16, WkT, bk, K16, MTOT, DD, DD);
  gemm_f16<1><<<ggrid, 256, 0, stream>>>(X16, WvT, bv, Vt16, MTOT, DD, DD);

  attn_kernel<<<BQ * HH * (TT / 128), 256, 0, stream>>>(Q16, K16, Vt16, mask, ctx16);

  gemm_f16<2><<<ggrid, 256, 0, stream>>>(ctx16, WoT, bo, out, MTOT, DD, DD);
}
